// qkv_attn_35966056137256
// MI455X (gfx1250) — matrix operand order repaired and checked by reading, not compiled, not run on hardware
//
#include <hip/hip_runtime.h>
#include <hip/hip_bf16.h>
#include <math.h>

// ---------------------------------------------------------------------------
// Reference shapes: B=8, HW=4096, C=256, HEADS=8, HEADC=32.
// N_ROWS = 32768.  qkv row layout: 768 = [head][q(32) k(32) v(32)].
// ---------------------------------------------------------------------------
#define N_ROWS 32768
#define KDIM   256
#define QKV_C  768
#define LDT    72      // LDS stride in halves for a 64-wide K stage (+8 pad)

typedef __attribute__((ext_vector_type(16))) _Float16     v16h;
typedef __attribute__((ext_vector_type(8)))  float        v8f;
typedef __attribute__((ext_vector_type(8)))  unsigned int v8u;

union Frag { v8u u; v16h h; };
union H2   { unsigned int u; _Float16 h[2]; };
union H4   { uint2 u2; _Float16 h[4]; };

// ---------------------------------------------------------------------------
// One-time f32 -> f16 weight conversion (w_qkv, w1, w2; 1.3 MB total).
// ---------------------------------------------------------------------------
__global__ __launch_bounds__(256) void cvt_f32_f16(
    const float* __restrict__ in, _Float16* __restrict__ out, int n4)
{
    int i = blockIdx.x * 256 + threadIdx.x;
    if (i < n4) {
        float4 f = ((const float4*)in)[i];
        H4 cv;
        cv.h[0] = (_Float16)f.x; cv.h[1] = (_Float16)f.y;
        cv.h[2] = (_Float16)f.z; cv.h[3] = (_Float16)f.w;
        ((uint2*)out)[i] = cv.u2;
    }
}

// ---------------------------------------------------------------------------
// Tiled WMMA GEMM:  C[M,NC] = epilogue( A[M,256] @ Bh[256,NC] ),  Bh = f16.
// Block tile 128x64; 8 waves, each a 32x32 macro-tile = 2x2
// v_wmma_f32_16x16x32_f16.  K staged 64 at a time (27.6 KB LDS):
//   A: float4 global loads -> cvt -> ds_store_b64   (row-major, pad 8)
//   B: uint4 (8-half) global loads -> transposed b16 stores ([col][k])
// so every lane's fragment is two contiguous ds_load_b128 per operand,
// matching the CDNA5 A(16x32)/B(32x16) f16 VGPR layouts.
// MODE 0: +bias | MODE 1: +bias,GELU(erf) | MODE 2: +bias+aux residual
// ---------------------------------------------------------------------------
template <int NC, int MODE>
__global__ __launch_bounds__(256) void gemm_wmma_f16(
    const float* __restrict__ A, const _Float16* __restrict__ Bh,
    const float* __restrict__ bias, const float* __restrict__ aux,
    float* __restrict__ C)
{
    __shared__ _Float16 sA[128 * LDT];   // 18.4 KB
    __shared__ _Float16 sB[64 * LDT];    //  9.2 KB

    const int tid  = threadIdx.x;
    const int lane = tid & 31;
    const int wid  = tid >> 5;
    const int wm   = wid & 3;            // 4 waves along M
    const int wn   = wid >> 2;           // 2 waves along N
    const int m0   = blockIdx.x * 128;
    const int n0   = blockIdx.y * 64;

    v8f cc[2][2] = {};

    const int arow = lane & 15;
    const int aklo = (lane < 16) ? 0 : 8;    // A: lanes 0-15 hold K 0..7 / 16..23
    const int bklo = (lane < 16) ? 0 : 8;    // B: lanes 0-15 hold K 0..7 / 16..23

    for (int k0 = 0; k0 < KDIM; k0 += 64) {
        // ---- stage A tile (128 rows x 64 K) as f16: 8 float4 per thread ----
        #pragma unroll
        for (int j = 0; j < 8; ++j) {
            int g    = tid + j * 256;        // group of 4 floats
            int r    = g >> 4;               // 16 groups per row
            int kpos = (g & 15) << 2;
            float4 f = *(const float4*)(A + (size_t)(m0 + r) * KDIM + k0 + kpos);
            H4 cv;
            cv.h[0] = (_Float16)f.x; cv.h[1] = (_Float16)f.y;
            cv.h[2] = (_Float16)f.z; cv.h[3] = (_Float16)f.w;
            *(uint2*)(sA + r * LDT + kpos) = cv.u2;
        }
        // ---- stage B tile (64 K x 64 cols), transposed: 2 uint4 per thread --
        #pragma unroll
        for (int j = 0; j < 2; ++j) {
            int g  = tid + j * 256;          // group of 8 halves
            int kk = g >> 3;                 // 8 groups per K row
            int c0 = (g & 7) << 3;
            uint4 wv = *(const uint4*)(Bh + (size_t)(k0 + kk) * NC + n0 + c0);
            unsigned int ww[4] = {wv.x, wv.y, wv.z, wv.w};
            #pragma unroll
            for (int i = 0; i < 4; ++i) {
                H2 t; t.u = ww[i];
                sB[(c0 + 2 * i)     * LDT + kk] = t.h[0];
                sB[(c0 + 2 * i + 1) * LDT + kk] = t.h[1];
            }
        }
        __syncthreads();

        // ---- two K=32 WMMA steps over the staged 64-wide slab ----
        #pragma unroll
        for (int ks = 0; ks < 64; ks += 32) {
            Frag af[2], bf[2];
            #pragma unroll
            for (int tm = 0; tm < 2; ++tm) {
                int row = wm * 32 + tm * 16 + arow;
                uint4 q0 = *(const uint4*)(sA + row * LDT + ks + aklo);
                uint4 q1 = *(const uint4*)(sA + row * LDT + ks + 16 + aklo);
                af[tm].u[0] = q0.x; af[tm].u[1] = q0.y;
                af[tm].u[2] = q0.z; af[tm].u[3] = q0.w;
                af[tm].u[4] = q1.x; af[tm].u[5] = q1.y;
                af[tm].u[6] = q1.z; af[tm].u[7] = q1.w;
            }
            #pragma unroll
            for (int tn = 0; tn < 2; ++tn) {
                int col = wn * 32 + tn * 16 + arow;
                uint4 q0 = *(const uint4*)(sB + col * LDT + ks + bklo);
                uint4 q1 = *(const uint4*)(sB + col * LDT + ks + 16 + bklo);
                bf[tn].u[0] = q0.x; bf[tn].u[1] = q0.y;
                bf[tn].u[2] = q0.z; bf[tn].u[3] = q0.w;
                bf[tn].u[4] = q1.x; bf[tn].u[5] = q1.y;
                bf[tn].u[6] = q1.z; bf[tn].u[7] = q1.w;
            }
            #pragma unroll
            for (int tm = 0; tm < 2; ++tm)
                #pragma unroll
                for (int tn = 0; tn < 2; ++tn)
                    cc[tm][tn] = __builtin_amdgcn_wmma_f32_16x16x32_f16(
                        false, af[tm].h, false, bf[tn].h,
                        (short)0, cc[tm][tn], false, false);
        }
        __syncthreads();
    }

    // ---- epilogue: C 16x16 f32 layout c[r] -> (row r or 8+r, col lane%16) ---
    #pragma unroll
    for (int tm = 0; tm < 2; ++tm) {
        #pragma unroll
        for (int tn = 0; tn < 2; ++tn) {
            int col  = n0 + wn * 32 + tn * 16 + (lane & 15);
            float bv = bias[col];
            #pragma unroll
            for (int r = 0; r < 8; ++r) {
                int row = m0 + wm * 32 + tm * 16 + ((lane < 16) ? r : r + 8);
                float v = cc[tm][tn][r] + bv;
                float o;
                if (MODE == 1) {
                    o = 0.5f * v * (1.0f + erff(v * 0.70710678118654752f));
                } else if (MODE == 2) {
                    o = v + aux[(size_t)row * 256 + col];
                } else {
                    o = v;
                }
                C[(size_t)row * NC + col] = o;
            }
        }
    }
}

// ---------------------------------------------------------------------------
// LN(k), LN(v) + kv = (k_ln^T v_ln)/HW per (b,h).  One block per (b,h);
// each wave32 owns one row (lane = channel), LN via shuffle reductions,
// normalized rows staged in LDS, each thread accumulates 4 kv entries.
// ---------------------------------------------------------------------------
__global__ __launch_bounds__(256) void ln_kv_kernel(
    const float* __restrict__ qkv,
    const float* __restrict__ kg, const float* __restrict__ kb,
    const float* __restrict__ vg, const float* __restrict__ vb,
    float* __restrict__ kvout)
{
    const int bh = blockIdx.x;
    const int b  = bh >> 3, h = bh & 7;
    const int tid  = threadIdx.x;
    const int wave = tid >> 5;
    const int lane = tid & 31;

    __shared__ float lk[8 * 32];
    __shared__ float lv[8 * 32];

    const float gk = kg[lane], bk = kb[lane];
    const float gv = vg[lane], bv = vb[lane];

    const float* base = qkv + (size_t)b * 4096 * QKV_C + h * 96 + 32;
    float acc[4] = {0.f, 0.f, 0.f, 0.f};

    for (int nc = 0; nc < 4096; nc += 8) {
        const float* pr = base + (size_t)(nc + wave) * QKV_C;
        float kx = pr[lane];
        float vx = pr[32 + lane];

        float s = kx;
        #pragma unroll
        for (int m = 16; m > 0; m >>= 1) s += __shfl_xor(s, m, 32);
        float d  = kx - s * (1.0f / 32.0f);
        float s2 = d * d;
        #pragma unroll
        for (int m = 16; m > 0; m >>= 1) s2 += __shfl_xor(s2, m, 32);
        float kn = d * rsqrtf(s2 * (1.0f / 32.0f) + 1e-5f) * gk + bk;

        float t = vx;
        #pragma unroll
        for (int m = 16; m > 0; m >>= 1) t += __shfl_xor(t, m, 32);
        float dv = vx - t * (1.0f / 32.0f);
        float t2 = dv * dv;
        #pragma unroll
        for (int m = 16; m > 0; m >>= 1) t2 += __shfl_xor(t2, m, 32);
        float vn = dv * rsqrtf(t2 * (1.0f / 32.0f) + 1e-5f) * gv + bv;

        __syncthreads();
        lk[wave * 32 + lane] = kn;
        lv[wave * 32 + lane] = vn;
        __syncthreads();

        #pragma unroll
        for (int r = 0; r < 8; ++r) {
            float ve = lv[r * 32 + lane];
            #pragma unroll
            for (int j = 0; j < 4; ++j)
                acc[j] += lk[r * 32 + wave + 8 * j] * ve;
        }
    }

    #pragma unroll
    for (int j = 0; j < 4; ++j)
        kvout[(size_t)bh * 1024 + (wave + 8 * j) * 32 + lane] =
            acc[j] * (1.0f / 4096.0f);
}

// ---------------------------------------------------------------------------
// ret = q @ kv + x.  One block = 32 rows; kv for the batch element in LDS.
// ---------------------------------------------------------------------------
__global__ __launch_bounds__(256) void q_apply_kernel(
    const float* __restrict__ qkv, const float* __restrict__ kv,
    const float* __restrict__ x, float* __restrict__ ret)
{
    const int row0 = blockIdx.x * 32;
    const int b    = row0 >> 12;
    const int tid  = threadIdx.x;
    const int head = tid >> 5, e = tid & 31;

    __shared__ float skv[8192];
    __shared__ float sq[256];

    for (int i = tid; i < 8192; i += 256)
        skv[i] = kv[(size_t)b * 8192 + i];
    __syncthreads();

    const float* kvh = skv + head * 1024;
    for (int r = 0; r < 32; ++r) {
        size_t n = (size_t)row0 + r;
        sq[tid] = qkv[n * QKV_C + head * 96 + e];
        __syncthreads();
        const float* qh = sq + head * 32;
        float acc = 0.f;
        #pragma unroll
        for (int d = 0; d < 32; ++d)
            acc = fmaf(qh[d], kvh[d * 32 + e], acc);
        ret[n * 256 + tid] = acc + x[n * 256 + tid];
        __syncthreads();
    }
}

// ---------------------------------------------------------------------------
// Launch: cvt weights -> qkv GEMM -> LN+kv -> q@kv+res -> MLP1 -> MLP2
// ws: [qkv 96MB][ret 32MB][kv 256KB][wqkv_h 384KB][w1_h 128KB][w2_h 128KB]
// MLP hidden reuses the qkv region (q consumed before overwrite).
// ---------------------------------------------------------------------------
extern "C" void kernel_launch(void* const* d_in, const int* in_sizes, int n_in,
                              void* d_out, int out_size, void* d_ws, size_t ws_size,
                              hipStream_t stream)
{
    const float* x     = (const float*)d_in[0];
    const float* w_qkv = (const float*)d_in[1];
    const float* b_qkv = (const float*)d_in[2];
    const float* kln_g = (const float*)d_in[3];
    const float* kln_b = (const float*)d_in[4];
    const float* vln_g = (const float*)d_in[5];
    const float* vln_b = (const float*)d_in[6];
    const float* w1    = (const float*)d_in[7];
    const float* b1    = (const float*)d_in[8];
    const float* w2    = (const float*)d_in[9];
    const float* b2    = (const float*)d_in[10];
    float*       out   = (float*)d_out;

    char* ws = (char*)d_ws;
    size_t off = 0;
    float* qkvb = (float*)(ws + off); off += (size_t)N_ROWS * QKV_C * 4;   // 96MB
    float* retb = (float*)(ws + off); off += (size_t)N_ROWS * 256 * 4;     // 32MB
    float* kvb  = (float*)(ws + off); off += (size_t)64 * 1024 * 4;        // 256KB
    _Float16* wqkv_h = (_Float16*)(ws + off); off += (size_t)KDIM * QKV_C * 2;
    _Float16* w1_h   = (_Float16*)(ws + off); off += (size_t)KDIM * 256 * 2;
    _Float16* w2_h   = (_Float16*)(ws + off); off += (size_t)KDIM * 256 * 2;
    float* hbuf = qkvb;   // reuse qkv region for MLP hidden

    dim3 blk(256);

    // 0) weights -> f16
    cvt_f32_f16<<<(KDIM * QKV_C / 4 + 255) / 256, blk, 0, stream>>>(
        w_qkv, wqkv_h, KDIM * QKV_C / 4);
    cvt_f32_f16<<<(KDIM * 256 / 4 + 255) / 256, blk, 0, stream>>>(
        w1, w1_h, KDIM * 256 / 4);
    cvt_f32_f16<<<(KDIM * 256 / 4 + 255) / 256, blk, 0, stream>>>(
        w2, w2_h, KDIM * 256 / 4);

    // 1) qkv = x @ w_qkv + b_qkv
    gemm_wmma_f16<QKV_C, 0><<<dim3(N_ROWS / 128, QKV_C / 64), blk, 0, stream>>>(
        x, wqkv_h, b_qkv, nullptr, qkvb);

    // 2) LN(k), LN(v), kv = k^T v / HW
    ln_kv_kernel<<<64, blk, 0, stream>>>(qkvb, kln_g, kln_b, vln_g, vln_b, kvb);

    // 3) ret = q @ kv + x
    q_apply_kernel<<<N_ROWS / 32, blk, 0, stream>>>(qkvb, kvb, x, retb);

    // 4) h = gelu(ret @ w1 + b1)
    gemm_wmma_f16<256, 1><<<dim3(N_ROWS / 128, 256 / 64), blk, 0, stream>>>(
        retb, w1_h, b1, nullptr, hbuf);

    // 5) out = h @ w2 + b2 + x
    gemm_wmma_f16<256, 2><<<dim3(N_ROWS / 128, 256 / 64), blk, 0, stream>>>(
        hbuf, w2_h, b2, x, out);
}
